// BiMambaBlock2_67430986547338
// MI455X (gfx1250) — matrix operand order repaired and checked by reading, not compiled, not run on hardware
//
#include <hip/hip_runtime.h>
#include <hip/hip_bf16.h>
#include <math.h>

// ---------------- sizes ----------------
#define BATCH   2
#define LSEQ    4096
#define DM      512      // D_MODEL
#define DI      1024     // D_INNER
#define NH      16       // NHEADS
#define HD      64       // HEADDIM
#define DSTATE  64
#define CD      1152     // CONV_DIM
#define DP      2192     // D_PROJ
#define ROWS    (BATCH*LSEQ)   // 8192
#define FFH     2048
#define EPSV    1e-5f

typedef _Float16 f16;
typedef __attribute__((ext_vector_type(16))) _Float16 v16h;
typedef __attribute__((ext_vector_type(8)))  _Float16 v8h;
typedef __attribute__((ext_vector_type(8)))  float    v8f;

// ---------------------------------------------------------------------------
// Weight convert + transpose: src f32 [K x N] row-major -> dst f16 [N x K]
// ---------------------------------------------------------------------------
__global__ void convert_transpose(const float* __restrict__ src, f16* __restrict__ dst,
                                  int K, int N) {
  size_t i = (size_t)blockIdx.x * blockDim.x + threadIdx.x;
  size_t total = (size_t)K * N;
  if (i < total) {
    int k = (int)(i / N);
    int n = (int)(i % N);
    dst[(size_t)n * K + k] = (f16)src[i];
  }
}

// ---------------------------------------------------------------------------
// LayerNorm over 512 cols -> f16
// ---------------------------------------------------------------------------
__global__ void layernorm_to_f16(const float* __restrict__ x, const float* __restrict__ w,
                                 const float* __restrict__ b, f16* __restrict__ out) {
  int row = blockIdx.x;                 // 8192 rows
  int tid = threadIdx.x;                // 256
  const float* xr = x + (size_t)row * DM;
  float v0 = xr[tid], v1 = xr[tid + 256];
  __shared__ float red[256];
  red[tid] = v0 + v1; __syncthreads();
  for (int off = 128; off > 0; off >>= 1) { if (tid < off) red[tid] += red[tid + off]; __syncthreads(); }
  float mu = red[0] * (1.0f / DM); __syncthreads();
  float d0 = v0 - mu, d1 = v1 - mu;
  red[tid] = d0 * d0 + d1 * d1; __syncthreads();
  for (int off = 128; off > 0; off >>= 1) { if (tid < off) red[tid] += red[tid + off]; __syncthreads(); }
  float rstd = rsqrtf(red[0] * (1.0f / DM) + EPSV);
  f16* orow = out + (size_t)row * DM;
  orow[tid]       = (f16)(d0 * rstd * w[tid]       + b[tid]);
  orow[tid + 256] = (f16)(d1 * rstd * w[tid + 256] + b[tid + 256]);
}

// ---------------------------------------------------------------------------
// fwd[row] + flip(bwd)[row] -> LayerNorm -> f16
// ---------------------------------------------------------------------------
__global__ void add_layernorm_to_f16(const float* __restrict__ fwd, const float* __restrict__ bwd,
                                     const float* __restrict__ w, const float* __restrict__ bb,
                                     f16* __restrict__ out) {
  int row = blockIdx.x;                       // b*L + t
  int bi = row >> 12, t = row & (LSEQ - 1);
  size_t r2 = ((size_t)bi << 12) + (LSEQ - 1 - t);
  int tid = threadIdx.x;
  float v0 = fwd[(size_t)row * DM + tid]       + bwd[r2 * DM + tid];
  float v1 = fwd[(size_t)row * DM + tid + 256] + bwd[r2 * DM + tid + 256];
  __shared__ float red[256];
  red[tid] = v0 + v1; __syncthreads();
  for (int off = 128; off > 0; off >>= 1) { if (tid < off) red[tid] += red[tid + off]; __syncthreads(); }
  float mu = red[0] * (1.0f / DM); __syncthreads();
  float d0 = v0 - mu, d1 = v1 - mu;
  red[tid] = d0 * d0 + d1 * d1; __syncthreads();
  for (int off = 128; off > 0; off >>= 1) { if (tid < off) red[tid] += red[tid + off]; __syncthreads(); }
  float rstd = rsqrtf(red[0] * (1.0f / DM) + EPSV);
  f16* orow = out + (size_t)row * DM;
  orow[tid]       = (f16)(d0 * rstd * w[tid]       + bb[tid]);
  orow[tid + 256] = (f16)(d1 * rstd * w[tid + 256] + bb[tid + 256]);
}

// ---------------------------------------------------------------------------
// A-fragment loader (contiguous k = 16h + s, matching B): two b128 loads.
// ---------------------------------------------------------------------------
__device__ __forceinline__ v16h loadA(const f16* __restrict__ ap) {
  v8h a0 = *(const v8h*)ap;
  v8h a1 = *(const v8h*)(ap + 8);
  return __builtin_shufflevector(a0, a1, 0,1,2,3,4,5,6,7,8,9,10,11,12,13,14,15);
}

// ---------------------------------------------------------------------------
// WMMA GEMM: C[MxN] = A[MxK] * B[KxN], BT given as [N x K] f16.
// One wave per 64x16 output block: 4 accumulators down M reuse the B (weight)
// fragment 4x. K-loop unrolled by 2 with explicit ping-pong fragment
// registers: load pong(k+32) -> WMMA(ping) -> load ping(k+64) -> WMMA(pong).
// No cur<-nxt register copies, and every WMMA group's operands were issued a
// half-iteration earlier (staggered s_wait_loadcnt, loads in flight).
// Requires K % 64 == 0 (true for 512 / 1024 / 2048 here).
// EPI: 0 = store f32; 1 = +bias, exact GELU, store f16; 2 = +bias +residual f32.
// ---------------------------------------------------------------------------
template <int EPI>
__global__ void gemm_wmma(const f16* __restrict__ A, const f16* __restrict__ BT,
                          float* __restrict__ Cf, f16* __restrict__ Ch,
                          const float* __restrict__ bias, const float* __restrict__ resid,
                          int M, int N, int K) {
  const int lane = threadIdx.x & 31;
  const int wid  = blockIdx.x * (blockDim.x >> 5) + (threadIdx.x >> 5);
  const int ntiles = N >> 4;
  const int mblk = wid / ntiles;           // 64-row block index
  const int nt   = wid - mblk * ntiles;
  if (mblk * 64 >= M) return;
  const int lr = lane & 15;
  const f16* ap0 = A + (size_t)(mblk * 64 + lr) * K + ((lane & 16) ? 16 : 0);
  const size_t mstride = (size_t)16 * K;
  const f16* bp = BT + (size_t)(nt * 16 + lr) * K + ((lane & 16) ? 16 : 0);
  v8f acc[4] = {};
  // ping registers: k = 0
  v16h b0 = *(const v16h*)bp;
  v16h a0[4];
#pragma unroll
  for (int m = 0; m < 4; m++) a0[m] = loadA(ap0 + m * mstride);
  for (int k0 = 0; k0 < K; k0 += 64) {
    // pong registers: k0 + 32
    v16h b1 = *(const v16h*)(bp + k0 + 32);
    v16h a1[4];
#pragma unroll
    for (int m = 0; m < 4; m++) a1[m] = loadA(ap0 + m * mstride + k0 + 32);
    __builtin_prefetch(bp + k0 + 256, 0, 0);
#pragma unroll
    for (int m = 0; m < 4; m++) {
      acc[m] = __builtin_amdgcn_wmma_f32_16x16x32_f16(false, a0[m], false, b0,
                                                      (short)0, acc[m], false, false);
    }
    if (k0 + 64 < K) {
      // ping registers: k0 + 64
      b0 = *(const v16h*)(bp + k0 + 64);
#pragma unroll
      for (int m = 0; m < 4; m++) a0[m] = loadA(ap0 + m * mstride + k0 + 64);
    }
#pragma unroll
    for (int m = 0; m < 4; m++) {
      acc[m] = __builtin_amdgcn_wmma_f32_16x16x32_f16(false, a1[m], false, b1,
                                                      (short)0, acc[m], false, false);
    }
  }
  const int col = nt * 16 + lr;
#pragma unroll
  for (int m = 0; m < 4; m++) {
    const int row = mblk * 64 + m * 16 + ((lane & 16) ? 8 : 0);
#pragma unroll
    for (int r = 0; r < 8; r++) {
      float v = acc[m][r];
      size_t idx = (size_t)(row + r) * N + col;
      if constexpr (EPI == 0) {
        Cf[idx] = v;
      } else if constexpr (EPI == 1) {
        v += bias[col];
        v = 0.5f * v * (1.0f + erff(v * 0.70710678118654752f));   // exact GELU
        Ch[idx] = (f16)v;
      } else {
        Cf[idx] = v + bias[col] + resid[idx];
      }
    }
  }
}

// ---------------------------------------------------------------------------
// Depthwise conv (w=4) + silu over xBC, plus dt -> softplus / dA. Direction d
// reads zxbcdt rows flipped (bwd stream never re-runs the in-proj GEMM).
// grid = 2*B*L blocks, 256 threads.
// ---------------------------------------------------------------------------
__global__ void conv_silu_split(const float* __restrict__ zx, const float* __restrict__ cw,
                                const float* __restrict__ cb, const float* __restrict__ dt_bias,
                                const float* __restrict__ A_log,
                                float* __restrict__ xh, float* __restrict__ Bc,
                                float* __restrict__ Cc, float* __restrict__ dAo,
                                float* __restrict__ dto) {
  int idx = blockIdx.x;
  int t = idx & (LSEQ - 1);
  int b = (idx >> 12) & 1;
  int d = idx >> 13;
  size_t orow = ((size_t)(d * BATCH + b)) * LSEQ + t;
  for (int c = threadIdx.x; c < CD; c += blockDim.x) {
    float acc = cb[c];
#pragma unroll
    for (int j = 0; j < 4; j++) {
      int tt = t - 3 + j;
      if (tt >= 0) {
        int pos = d ? (LSEQ - 1 - tt) : tt;
        acc += cw[j * CD + c] * zx[((size_t)b * LSEQ + pos) * DP + DI + c];
      }
    }
    float s = acc / (1.0f + __expf(-acc));      // silu
    if (c < DI)               xh[orow * DI + c] = s;
    else if (c < DI + DSTATE) Bc[orow * DSTATE + (c - DI)] = s;
    else                      Cc[orow * DSTATE + (c - DI - DSTATE)] = s;
  }
  if (threadIdx.x < NH) {
    int h = threadIdx.x;
    int pos = d ? (LSEQ - 1 - t) : t;
    float draw = zx[((size_t)b * LSEQ + pos) * DP + (DP - NH) + h] + dt_bias[h];
    float sp = (draw > 20.0f) ? draw : log1pf(__expf(draw));   // softplus
    dto[orow * NH + h] = sp;
    dAo[orow * NH + h] = __expf(-__expf(A_log[h]) * sp);
  }
}

// ---------------------------------------------------------------------------
// Selective scan. One block per (dir,batch,head) stream = 64 blocks.
// 256 threads: thread (p = tid/4, 16 states n = (tid%4)*16 .. +15).
// Double-buffered LDS pipeline: while chunk c is computed, chunk c+1 streams
// in via GLOBAL_LOAD_ASYNC_TO_LDS_B128 (ASYNCcnt). One s_wait_asynccnt +
// barrier per chunk. y reduced with wave32 shuffles.
// ---------------------------------------------------------------------------
__global__ void __launch_bounds__(256) scan_kernel(
    const float* __restrict__ xh, const float* __restrict__ Bc,
    const float* __restrict__ Cc, const float* __restrict__ dA,
    const float* __restrict__ dt, const float* __restrict__ D_h,
    float* __restrict__ y) {
  int bid = blockIdx.x;
  int h = bid & (NH - 1);
  int b = (bid >> 4) & 1;
  int d = bid >> 5;
  int tid = threadIdx.x;
  int p  = tid >> 2;
  int ng = (tid & 3) * 16;
  __shared__ __align__(16) float sB[2][64 * DSTATE];
  __shared__ __align__(16) float sC[2][64 * DSTATE];
  __shared__ __align__(16) float sX[2][64 * HD];
  __shared__ float sA[2][64];
  __shared__ float sW[2][64];
  float hst[16];
#pragma unroll
  for (int i = 0; i < 16; i++) hst[i] = 0.0f;
  float Dv = D_h[h];
  size_t base = ((size_t)(d * BATCH + b)) * LSEQ;

  // flat->LDS aperture: the low 32 bits of the address are the LDS offset, so truncation gives it
  auto issue_chunk = [&](int c0, int buf) {
    const float* bsrc = Bc + (base + c0) * DSTATE;   // contiguous 16KB
    const float* csrc = Cc + (base + c0) * DSTATE;   // contiguous 16KB
    unsigned lB = (unsigned)(uintptr_t)&sB[buf][0];
    unsigned lC = (unsigned)(uintptr_t)&sC[buf][0];
    unsigned lX = (unsigned)(uintptr_t)&sX[buf][0];
#pragma unroll
    for (int j = 0; j < 4; j++) {
      int idx = tid + j * 256;            // float4 index 0..1023
      unsigned long long gb = (unsigned long long)(uintptr_t)(bsrc + (size_t)idx * 4);
      unsigned long long gc = (unsigned long long)(uintptr_t)(csrc + (size_t)idx * 4);
      asm volatile("global_load_async_to_lds_b128 %0, %1, off"
                   :: "v"(lB + (unsigned)idx * 16u), "v"(gb) : "memory");
      asm volatile("global_load_async_to_lds_b128 %0, %1, off"
                   :: "v"(lC + (unsigned)idx * 16u), "v"(gc) : "memory");
    }
    {
      int tt = tid >> 2, q = tid & 3;     // xh slice: 64 rows x 64B, one b128/lane
      unsigned long long gx =
          (unsigned long long)(uintptr_t)(xh + (base + c0 + tt) * DI + h * HD + q * 16);
      asm volatile("global_load_async_to_lds_b128 %0, %1, off"
                   :: "v"(lX + (unsigned)(tt * HD + q * 16) * 4u), "v"(gx) : "memory");
    }
    for (int i = tid; i < 64; i += 256) {
      size_t row = base + c0 + i;
      sA[buf][i] = dA[row * NH + h];
      sW[buf][i] = dt[row * NH + h];
    }
  };

  issue_chunk(0, 0);
  const int NCHUNK = LSEQ / 64;
  for (int c = 0; c < NCHUNK; c++) {
    const int buf = c & 1;
    asm volatile("s_wait_asynccnt 0x0" ::: "memory");   // chunk c copies done (this wave)
    __syncthreads();                                    // all waves' copies visible
    if (c + 1 < NCHUNK) issue_chunk((c + 1) * 64, buf ^ 1);   // overlap with compute
    const int c0 = c * 64;
    for (int tt = 0; tt < 64; tt++) {
      float a = sA[buf][tt];
      float xp = sX[buf][tt * HD + p];
      float coef = sW[buf][tt] * xp;
      float yv = 0.0f;
#pragma unroll
      for (int i = 0; i < 16; i++) {
        hst[i] = hst[i] * a + coef * sB[buf][tt * DSTATE + ng + i];
        yv += hst[i] * sC[buf][tt * DSTATE + ng + i];
      }
      yv += __shfl_xor(yv, 1, 32);
      yv += __shfl_xor(yv, 2, 32);
      if ((tid & 3) == 0) {
        size_t row = base + c0 + tt;
        y[row * DI + h * HD + p] = yv + Dv * xp;
      }
    }
  }
}

// ---------------------------------------------------------------------------
// g = y * silu(z), RMS-norm over 1024, * mnorm_w -> f16. z read flipped for bwd.
// grid = 2*B*L = 16384 blocks, 256 threads, 4 elems/thread.
// ---------------------------------------------------------------------------
__global__ void gate_rmsnorm(const float* __restrict__ y, const float* __restrict__ zx,
                             const float* __restrict__ mw, f16* __restrict__ g) {
  int row = blockIdx.x;
  int d = row >> 13;
  int b = (row >> 12) & 1;
  int t = row & (LSEQ - 1);
  size_t zrow = ((size_t)b * LSEQ + (d ? (LSEQ - 1 - t) : t)) * DP;
  int tid = threadIdx.x;
  float vals[4];
  float ss = 0.0f;
#pragma unroll
  for (int i = 0; i < 4; i++) {
    int c = tid + i * 256;
    float z = zx[zrow + c];
    float gv = y[(size_t)row * DI + c] * (z / (1.0f + __expf(-z)));
    vals[i] = gv;
    ss += gv * gv;
  }
  __shared__ float red[256];
  red[tid] = ss; __syncthreads();
  for (int off = 128; off > 0; off >>= 1) { if (tid < off) red[tid] += red[tid + off]; __syncthreads(); }
  float rr = rsqrtf(red[0] * (1.0f / DI) + EPSV);
#pragma unroll
  for (int i = 0; i < 4; i++) {
    int c = tid + i * 256;
    g[(size_t)row * DI + c] = (f16)(vals[i] * rr * mw[c]);
  }
}

// ---------------------------------------------------------------------------
static inline int gemm_blocks(int M, int N) {
  long long waves = (long long)(M / 64) * (N / 16);
  return (int)((waves + 7) / 8);          // 8 waves / 256-thread block
}

extern "C" void kernel_launch(void* const* d_in, const int* in_sizes, int n_in,
                              void* d_out, int out_size, void* d_ws, size_t ws_size,
                              hipStream_t stream) {
  const float* x       = (const float*)d_in[0];
  const float* nin_w   = (const float*)d_in[1];
  const float* nin_b   = (const float*)d_in[2];
  const float* nout_w  = (const float*)d_in[3];
  const float* nout_b  = (const float*)d_in[4];
  const float* in_w    = (const float*)d_in[5];
  const float* conv_w  = (const float*)d_in[6];
  const float* conv_b  = (const float*)d_in[7];
  const float* dt_bias = (const float*)d_in[8];
  const float* A_log   = (const float*)d_in[9];
  const float* D_h     = (const float*)d_in[10];
  const float* mnorm_w = (const float*)d_in[11];
  const float* out_w   = (const float*)d_in[12];
  const float* ff_w1   = (const float*)d_in[13];
  const float* ff_b1   = (const float*)d_in[14];
  const float* ff_w2   = (const float*)d_in[15];
  const float* ff_b2   = (const float*)d_in[16];
  float* out = (float*)d_out;

  char* ws = (char*)d_ws;
  auto alloc = [&](size_t bytes) -> char* {
    char* p = ws;
    ws += (bytes + 255) & ~(size_t)255;
    return p;
  };
  f16*   in_wT  = (f16*)  alloc((size_t)DP * DM * 2);
  f16*   out_wT = (f16*)  alloc((size_t)DM * DI * 2);
  f16*   ff1T   = (f16*)  alloc((size_t)FFH * DM * 2);
  f16*   ff2T   = (f16*)  alloc((size_t)DM * FFH * 2);
  f16*   xn16   = (f16*)  alloc((size_t)ROWS * DM * 2);
  float* zx     = (float*)alloc((size_t)ROWS * DP * 4);
  float* xh     = (float*)alloc((size_t)2 * ROWS * DI * 4);
  float* Bc     = (float*)alloc((size_t)2 * ROWS * DSTATE * 4);
  float* Cc     = (float*)alloc((size_t)2 * ROWS * DSTATE * 4);
  float* dAb    = (float*)alloc((size_t)2 * ROWS * NH * 4);
  float* dtb    = (float*)alloc((size_t)2 * ROWS * NH * 4);
  float* ybuf   = (float*)alloc((size_t)2 * ROWS * DI * 4);
  f16*   g16    = (f16*)  alloc((size_t)2 * ROWS * DI * 2);
  float* outd   = (float*)alloc((size_t)2 * ROWS * DM * 4);
  f16*   m16    = (f16*)  alloc((size_t)ROWS * DM * 2);
  f16*   ff116  = (f16*)  alloc((size_t)ROWS * FFH * 2);

  // weight convert + transpose (f32 KxN -> f16 NxK)
  {
    auto ct = [&](const float* s, f16* dst, int K, int N) {
      int total = K * N;
      convert_transpose<<<(total + 255) / 256, 256, 0, stream>>>(s, dst, K, N);
    };
    ct(in_w,  in_wT,  DM,  DP);
    ct(out_w, out_wT, DI,  DM);
    ct(ff_w1, ff1T,   DM,  FFH);
    ct(ff_w2, ff2T,   FFH, DM);
  }

  // 1) input layernorm
  layernorm_to_f16<<<ROWS, 256, 0, stream>>>(x, nin_w, nin_b, xn16);

  // 2) in-projection (shared by both directions: flip(x)@W == flip(x@W))
  gemm_wmma<0><<<gemm_blocks(ROWS, DP), 256, 0, stream>>>(
      xn16, in_wT, zx, nullptr, nullptr, nullptr, ROWS, DP, DM);

  // 3) conv + silu + split + dt/dA, both directions
  conv_silu_split<<<2 * ROWS, 256, 0, stream>>>(zx, conv_w, conv_b, dt_bias, A_log,
                                                xh, Bc, Cc, dAb, dtb);

  // 4) selective scan (64 independent streams)
  scan_kernel<<<2 * BATCH * NH, 256, 0, stream>>>(xh, Bc, Cc, dAb, dtb, D_h, ybuf);

  // 5) gating + RMS norm
  gate_rmsnorm<<<2 * ROWS, 256, 0, stream>>>(ybuf, zx, mnorm_w, g16);

  // 6) out-projection, both directions in one launch (contiguous M blocks)
  gemm_wmma<0><<<gemm_blocks(2 * ROWS, DM), 256, 0, stream>>>(
      g16, out_wT, outd, nullptr, nullptr, nullptr, 2 * ROWS, DM, DI);

  // 7) fwd + flip(bwd), output layernorm
  add_layernorm_to_f16<<<ROWS, 256, 0, stream>>>(outd, outd + (size_t)ROWS * DM,
                                                 nout_w, nout_b, m16);

  // 8) FFN up: +bias, exact GELU -> f16
  gemm_wmma<1><<<gemm_blocks(ROWS, FFH), 256, 0, stream>>>(
      m16, ff1T, nullptr, ff116, ff_b1, nullptr, ROWS, FFH, DM);

  // 9) FFN down: +bias +residual(x) -> f32 output
  gemm_wmma<2><<<gemm_blocks(ROWS, DM), 256, 0, stream>>>(
      ff116, ff2T, out, nullptr, ff_b2, x, ROWS, DM, FFH);
}
